// STPrivacy_5007931867386
// MI455X (gfx1250) — matrix operand order repaired and checked by reading, not compiled, not run on hardware
//
#include <hip/hip_runtime.h>
#include <math.h>

// ---------------------------------------------------------------------------
// STPrivacy-style pruned video ViT forward, CDNA5 (gfx1250, wave32, WMMA).
// Dense math: v_wmma_f32_16x16x32_bf16 with f32 accumulate, LDS-staged tiles.
// Master activations stay f32 in memory; operands converted to bf16 on stage.
// ---------------------------------------------------------------------------

typedef __attribute__((ext_vector_type(16))) __bf16 v16bf;
typedef __attribute__((ext_vector_type(8)))  __bf16 v8bf;
typedef __attribute__((ext_vector_type(8)))  float  v8f;

#define EP_BIAS 1
#define EP_GELU 2
#define EP_RES  4

#define BM 64      // block tile rows (A/C)
#define BN 128     // block tile cols (W rows / C cols)
#define BK 32      // k-step
#define PITCH 48   // LDS row pitch in bf16 (96B: 16B aligned, bank-rotating)

// ---------------------------------------------------------------------------
// Batched GEMM:  C[m,n] = alpha * sum_k A[m,k] * W[n,k] (+bias[n]) (+res[m,n])
//                (optional exact-erf GELU before residual add)
// A(m,k) at A + m*aR + k*aK ; W(n,k) at W + n*wR + k*wK. All strides int
// (every offset in this net fits in 31 bits) to avoid 64-bit VALU math.
// Block: 256 threads = 8 waves in a 4x2 grid; wave computes 16x64 of C.
// Per K-step: block stages A(64x32) and B(128x32) to LDS as bf16
// (float4 global loads + packed b128 LDS stores on interior tiles; clamped
// gather-then-mask scalars on edges), then each wave reads WMMA fragments:
//   A (16x32): lane l -> row m=l&15; k(e) = (l>>4)*16 + e
//              => two 16B LDS reads at k-offsets (l>>4)*16 and (l>>4)*16+8
//   B (32x16): lane l -> col n=l&15; k(e) = (l>>4)*16 + e  => one 32B read
//   C (16x16): lane l -> col n=l&15; row m = j + 8*(l>>4)
// ---------------------------------------------------------------------------
__global__ __launch_bounds__(256) void k_gemm_bf16(
    const float* __restrict__ A, int aR, int aK, int aB,
    const float* __restrict__ W, int wR, int wK, int wB,
    float* __restrict__ C, int cR, int cB,
    const float* __restrict__ bias,
    const float* __restrict__ res, int rR, int rB,
    int M, int Nout, int K, float alpha, int flags)
{
    __shared__ __bf16 smA[BM * PITCH];
    __shared__ __bf16 smB[BN * PITCH];

    const int tid  = threadIdx.x;
    const int lane = tid & 31;
    const int w    = tid >> 5;      // 0..7
    const int wm   = w & 3;         // wave row-group (16 rows each)
    const int wn   = w >> 2;        // wave col-group (64 cols each)
    const int lm   = lane & 15;
    const int lh   = lane >> 4;

    const int m0 = blockIdx.x * BM;
    const int n0 = blockIdx.y * BN;
    const int bz = blockIdx.z;
    A += bz * aB; W += bz * wB; C += bz * cB;
    if (flags & EP_RES) res += bz * rB;

    // staging assignments: A: 8 elems/thread, B: 16 elems/thread
    const int arow = tid >> 2;            // 0..63
    const int aseg = (tid & 3) * 8;       // 0,8,16,24
    const int brow = tid >> 1;            // 0..127
    const int bseg = (tid & 1) * 16;      // 0,16

    const int agrow = m0 + arow;
    const int bgrow = n0 + brow;
    const float* aBase = A + (agrow < M ? agrow : 0) * aR;       // clamped row
    const float* bBase = W + (bgrow < Nout ? bgrow : 0) * wR;

    v8f acc[4];
#pragma unroll
    for (int t = 0; t < 4; ++t) acc[t] = (v8f){0.f,0.f,0.f,0.f,0.f,0.f,0.f,0.f};

    for (int k0 = 0; k0 < K; k0 += BK) {
        // ---------------- stage A tile (one b128 LDS store) ----------------
        {
            v8bf pk;
            if ((agrow < M) & (k0 + BK <= K) & (aK == 1)) {
                const float4* src = reinterpret_cast<const float4*>(aBase + k0 + aseg);
                float4 f0 = src[0], f1 = src[1];
                pk[0] = (__bf16)f0.x; pk[1] = (__bf16)f0.y;
                pk[2] = (__bf16)f0.z; pk[3] = (__bf16)f0.w;
                pk[4] = (__bf16)f1.x; pk[5] = (__bf16)f1.y;
                pk[6] = (__bf16)f1.z; pk[7] = (__bf16)f1.w;
            } else {
                float tmp[8], mk[8];
#pragma unroll
                for (int j = 0; j < 8; ++j) {
                    int k = k0 + aseg + j;
                    int kc = (k < K) ? k : 0;
                    mk[j]  = ((agrow < M) && (k < K)) ? 1.0f : 0.0f;
                    tmp[j] = aBase[kc * aK];
                }
#pragma unroll
                for (int j = 0; j < 8; ++j) pk[j] = (__bf16)(tmp[j] * mk[j]);
            }
            *reinterpret_cast<v8bf*>(&smA[arow * PITCH + aseg]) = pk;
        }
        // ---------------- stage B tile (two b128 LDS stores) ----------------
        {
            v16bf pk;
            if ((bgrow < Nout) & (k0 + BK <= K) & (wK == 1)) {
                const float4* src = reinterpret_cast<const float4*>(bBase + k0 + bseg);
                float4 f0 = src[0], f1 = src[1], f2 = src[2], f3 = src[3];
                pk[0]  = (__bf16)f0.x; pk[1]  = (__bf16)f0.y;
                pk[2]  = (__bf16)f0.z; pk[3]  = (__bf16)f0.w;
                pk[4]  = (__bf16)f1.x; pk[5]  = (__bf16)f1.y;
                pk[6]  = (__bf16)f1.z; pk[7]  = (__bf16)f1.w;
                pk[8]  = (__bf16)f2.x; pk[9]  = (__bf16)f2.y;
                pk[10] = (__bf16)f2.z; pk[11] = (__bf16)f2.w;
                pk[12] = (__bf16)f3.x; pk[13] = (__bf16)f3.y;
                pk[14] = (__bf16)f3.z; pk[15] = (__bf16)f3.w;
            } else {
                float tmp[16], mk[16];
#pragma unroll
                for (int j = 0; j < 16; ++j) {
                    int k = k0 + bseg + j;
                    int kc = (k < K) ? k : 0;
                    mk[j]  = ((bgrow < Nout) && (k < K)) ? 1.0f : 0.0f;
                    tmp[j] = bBase[kc * wK];
                }
#pragma unroll
                for (int j = 0; j < 16; ++j) pk[j] = (__bf16)(tmp[j] * mk[j]);
            }
            *reinterpret_cast<v16bf*>(&smB[brow * PITCH + bseg]) = pk;
        }
        __syncthreads();

        // ---------------- fragments + WMMA ----------------
        v16bf afrag;
        {
            const int rbase = (wm * 16 + lm) * PITCH;
            v8bf alo = *reinterpret_cast<const v8bf*>(&smA[rbase + lh * 16]);
            v8bf ahi = *reinterpret_cast<const v8bf*>(&smA[rbase + lh * 16 + 8]);
#pragma unroll
            for (int j = 0; j < 8; ++j) { afrag[j] = alo[j]; afrag[j + 8] = ahi[j]; }
        }
#pragma unroll
        for (int t = 0; t < 4; ++t) {
            const v16bf bfrag = *reinterpret_cast<const v16bf*>(
                &smB[(wn * 64 + t * 16 + lm) * PITCH + lh * 16]);
            acc[t] = __builtin_amdgcn_wmma_f32_16x16x32_bf16(
                false, afrag, false, bfrag, (short)0, acc[t], false, false);
        }
        __syncthreads();
    }

    // ---------------- epilogue ----------------
#pragma unroll
    for (int t = 0; t < 4; ++t) {
        const int ncol = n0 + wn * 64 + t * 16 + lm;
        if (ncol >= Nout) continue;
        float bv = (flags & EP_BIAS) ? bias[ncol] : 0.0f;
#pragma unroll
        for (int j = 0; j < 8; ++j) {
            const int m = m0 + wm * 16 + lh * 8 + j;
            if (m >= M) continue;
            float v = acc[t][j] * alpha + bv;
            if (flags & EP_GELU) v = 0.5f * v * (1.0f + erff(v * 0.70710678118654752f));
            if (flags & EP_RES)  v += res[m * rR + ncol];
            C[m * cR + ncol] = v;
        }
    }
}

// ---------------------------------------------------------------------------
// Row LayerNorm over D=768 (eps = 1e-6). One block (256 threads) per row.
// ---------------------------------------------------------------------------
__global__ __launch_bounds__(256) void k_layernorm(
    const float* __restrict__ x, float* __restrict__ y,
    const float* __restrict__ g, const float* __restrict__ b)
{
    __shared__ float sm[256];
    const int row = blockIdx.x;
    const float* xr = x + (long long)row * 768;
    float* yr = y + (long long)row * 768;
    const int t = threadIdx.x;
    float v0 = xr[t], v1 = xr[t + 256], v2 = xr[t + 512];
    sm[t] = v0 + v1 + v2;
    __syncthreads();
    for (int o = 128; o > 0; o >>= 1) { if (t < o) sm[t] += sm[t + o]; __syncthreads(); }
    float mean = sm[0] * (1.0f / 768.0f);
    __syncthreads();
    float d0 = v0 - mean, d1 = v1 - mean, d2 = v2 - mean;
    sm[t] = d0 * d0 + d1 * d1 + d2 * d2;
    __syncthreads();
    for (int o = 128; o > 0; o >>= 1) { if (t < o) sm[t] += sm[t + o]; __syncthreads(); }
    float r = rsqrtf(sm[0] * (1.0f / 768.0f) + 1e-6f);
    yr[t]       = d0 * r * g[t]       + b[t];
    yr[t + 256] = d1 * r * g[t + 256] + b[t + 256];
    yr[t + 512] = d2 * r * g[t + 512] + b[t + 512];
}

// ---------------------------------------------------------------------------
// Policy-masked softmax over one score row (per-head buffer, in place):
//   ap[m] = p[m] + (1-p[m])*(m==n);  a = exp(s - max(s)) * ap
//   a = (a + EPS/N) / (sum(a) + EPS)
// grid = (Ntok, B), block = 128
// ---------------------------------------------------------------------------
__global__ __launch_bounds__(128) void k_attn_softmax(
    float* __restrict__ att, const float* __restrict__ pol, int Ntok)
{
    __shared__ float sm[128];
    const int n = blockIdx.x;
    const int b = blockIdx.y;
    float* row = att + ((long long)b * Ntok + n) * (long long)Ntok;
    const float* p = pol + (long long)b * Ntok;
    const int t = threadIdx.x;

    float mx = -3.402823e38f;
    for (int m = t; m < Ntok; m += 128) mx = fmaxf(mx, row[m]);
    sm[t] = mx; __syncthreads();
    for (int o = 64; o > 0; o >>= 1) { if (t < o) sm[t] = fmaxf(sm[t], sm[t + o]); __syncthreads(); }
    mx = sm[0]; __syncthreads();

    float s = 0.0f;
    for (int m = t; m < Ntok; m += 128) {
        float pv = p[m];
        float ap = pv + (1.0f - pv) * ((m == n) ? 1.0f : 0.0f);
        float e = expf(row[m] - mx) * ap;
        row[m] = e;
        s += e;
    }
    sm[t] = s; __syncthreads();
    for (int o = 64; o > 0; o >>= 1) { if (t < o) sm[t] += sm[t + o]; __syncthreads(); }
    s = sm[0];

    const float inv = 1.0f / (s + 1e-6f);
    const float add = 1e-6f / (float)Ntok;
    for (int m = t; m < Ntok; m += 128) row[m] = (row[m] + add) * inv;
}

// ---------------------------------------------------------------------------
// V transpose: vt[(b*768 + d)*392 + m] = qkv[(b*392 + m)*2304 + 1536 + d]
// Makes the a@v GEMM a unit-k-stride (fast path) GEMM.
// One block per (b,d) row of 392, 256 threads.
// ---------------------------------------------------------------------------
__global__ __launch_bounds__(256) void k_vtrans(
    const float* __restrict__ qkv, float* __restrict__ vt)
{
    const int bd = blockIdx.x;            // 0 .. 16*768-1
    const int bb = bd / 768;
    const int d  = bd - bb * 768;
    const float* src = qkv + ((long long)bb * 392) * 2304 + 1536 + d;
    float* dst = vt + (long long)bd * 392;
    for (int m = threadIdx.x; m < 392; m += 256)
        dst[m] = src[(long long)m * 2304];
}

// ---------------------------------------------------------------------------
// Patch gather: x(B,3,16,112,112) -> xp(B, N=392, K=1536)
// One block per token row (bn), 256 threads, 6 chunks of kk.
// tok = t*49 + ph*7 + pw ; kk = ((c*2 + ts)*16 + py)*16 + px ; frame f=t*2+ts
// ---------------------------------------------------------------------------
__global__ __launch_bounds__(256) void k_patch_gather(
    const float* __restrict__ x, float* __restrict__ xp)
{
    const int bn  = blockIdx.x;          // 0..6271
    const int b   = bn / 392;
    const int tok = bn - b * 392;
    const int tt = tok / 49; const int sp = tok - tt * 49;
    const int ph = sp / 7;   const int pw = sp - ph * 7;
    float* dst = xp + (long long)bn * 1536;
    const float* src = x + (long long)b * 3 * 16 * 112 * 112;
    for (int kk = threadIdx.x; kk < 1536; kk += 256) {
        int c  = kk >> 9;            // /512
        int r  = kk & 511;
        int ts = r >> 8;             // /256
        int r2 = r & 255;
        int py = r2 >> 4;
        int px = r2 & 15;
        int f = tt * 2 + ts;
        int rowi = ph * 16 + py;
        int coli = pw * 16 + px;
        dst[kk] = src[(((long long)c * 16 + f) * 112 + rowi) * 112 + coli];
    }
}

// pos add: one block per token row; pos period handled by scalar row mod
__global__ __launch_bounds__(256) void k_posadd(
    float* __restrict__ h, const float* __restrict__ pos)
{
    const int row = blockIdx.x;              // 0..6271
    const int prow = row % 392;
    float* hr = h + (long long)row * 768;
    const float* pr = pos + (long long)prow * 768;
    const int t = threadIdx.x;
    hr[t]       += pr[t];
    hr[t + 256] += pr[t + 256];
    hr[t + 512] += pr[t + 512];
}

__global__ __launch_bounds__(256) void k_fill(float* __restrict__ p, int n, float v)
{
    int i = blockIdx.x * 256 + threadIdx.x;
    if (i < n) p[i] = v;
}

// frame pooling: frame[bt,c] = sum_n h[bt,n,256+c]*pol / sum_n pol  (49 tokens)
__global__ __launch_bounds__(256) void k_frame_pool(
    const float* __restrict__ ph, const float* __restrict__ pol, float* __restrict__ frame)
{
    const int bt = blockIdx.x;       // 0..127  (B*T)
    const int c  = threadIdx.x;      // 0..255
    const float* hb = ph + (long long)bt * 49 * 768;
    const float* pb = pol + (long long)bt * 49;
    float s = 0.0f, ps = 0.0f;
    for (int n = 0; n < 49; ++n) {
        float pv = pb[n];
        s  += hb[(long long)n * 768 + 256 + c] * pv;
        ps += pv;
    }
    frame[(long long)bt * 256 + c] = s / ps;
}

// video pooling over all N=392 tokens, channels 512..767
__global__ __launch_bounds__(256) void k_video_pool(
    const float* __restrict__ ph, const float* __restrict__ pol, float* __restrict__ video)
{
    const int b = blockIdx.x;
    const int c = threadIdx.x;
    const float* hb = ph + (long long)b * 392 * 768;
    const float* pb = pol + (long long)b * 392;
    float s = 0.0f, ps = 0.0f;
    for (int n = 0; n < 392; ++n) {
        float pv = pb[n];
        s  += hb[(long long)n * 768 + 512 + c] * pv;
        ps += pv;
    }
    video[(long long)b * 256 + c] = s / ps;
}

// feat = concat(local, frame-broadcast, video-broadcast); one block per token
__global__ __launch_bounds__(256) void k_feat(
    const float* __restrict__ ph, const float* __restrict__ frame,
    const float* __restrict__ video, float* __restrict__ feat)
{
    const int bn = blockIdx.x;               // 0..6271
    const int b  = bn / 392;
    const int t  = (bn - b * 392) / 49;
    const int c  = threadIdx.x;              // 0..255
    const float* loc = ph + (long long)bn * 768;
    float* dst = feat + (long long)bn * 768;
    dst[c]       = loc[c];
    dst[c + 256] = frame[((long long)b * 8 + t) * 256 + c];
    dst[c + 512] = video[(long long)b * 256 + c];
}

// Straight-through gumbel decision (forward value):
//   keep iff z0 + g0 >= z1 + g1  (log_softmax shift is common to both logits)
__global__ __launch_bounds__(256) void k_decide(
    const float* __restrict__ zc, const float* __restrict__ g,
    float* __restrict__ pol, int n)
{
    int i = blockIdx.x * 256 + threadIdx.x;
    if (i >= n) return;
    float s0 = zc[2 * i]     + g[2 * i];
    float s1 = zc[2 * i + 1] + g[2 * i + 1];
    if (!(s0 >= s1)) pol[i] = 0.0f;
}

// ---------------------------------------------------------------------------
// Host-side launcher
// ---------------------------------------------------------------------------
static inline void gemm(hipStream_t st,
    const float* A, int aR, int aK, int aB,
    const float* W, int wR, int wK, int wB,
    float* C, int cR, int cB,
    const float* bias, const float* res, int rR, int rB,
    int M, int Nout, int K, float alpha, int flags, int batch)
{
    dim3 g((unsigned)((M + BM - 1) / BM), (unsigned)((Nout + BN - 1) / BN), (unsigned)batch);
    k_gemm_bf16<<<g, dim3(256), 0, st>>>(A, aR, aK, aB, W, wR, wK, wB,
                                         C, cR, cB, bias, res, rR, rB,
                                         M, Nout, K, alpha, flags);
}

extern "C" void kernel_launch(void* const* d_in, const int* in_sizes, int n_in,
                              void* d_out, int out_size, void* d_ws, size_t ws_size,
                              hipStream_t stream)
{
    (void)in_sizes; (void)n_in; (void)out_size; (void)ws_size;

    // --- unpack inputs (setup_inputs() dict insertion order, recursively) ---
    const float* X = (const float*)d_in[0];
    int p = 1;
    const float* patch_w = (const float*)d_in[p++];   // (768, 1536)
    const float* patch_b = (const float*)d_in[p++];   // (768)
    const float* pos     = (const float*)d_in[p++];   // (1, 392, 768)
    struct Blk { const float *ln1_w,*ln1_b,*qkv_w,*qkv_b,*proj_w,*proj_b,
                             *ln2_w,*ln2_b,*fc1_w,*fc1_b,*fc2_w,*fc2_b; } bl[12];
    for (int i = 0; i < 12; ++i) {
        bl[i].ln1_w = (const float*)d_in[p++]; bl[i].ln1_b = (const float*)d_in[p++];
        bl[i].qkv_w = (const float*)d_in[p++]; bl[i].qkv_b = (const float*)d_in[p++];
        bl[i].proj_w= (const float*)d_in[p++]; bl[i].proj_b= (const float*)d_in[p++];
        bl[i].ln2_w = (const float*)d_in[p++]; bl[i].ln2_b = (const float*)d_in[p++];
        bl[i].fc1_w = (const float*)d_in[p++]; bl[i].fc1_b = (const float*)d_in[p++];
        bl[i].fc2_w = (const float*)d_in[p++]; bl[i].fc2_b = (const float*)d_in[p++];
    }
    struct Prd { const float *ln_w,*ln_b,*in_w,*in_b,*a_w,*a_b,*b_w,*b_b,*c_w,*c_b; } pr[3];
    for (int i = 0; i < 3; ++i) {
        pr[i].ln_w = (const float*)d_in[p++]; pr[i].ln_b = (const float*)d_in[p++];
        pr[i].in_w = (const float*)d_in[p++]; pr[i].in_b = (const float*)d_in[p++];
        pr[i].a_w  = (const float*)d_in[p++]; pr[i].a_b  = (const float*)d_in[p++];
        pr[i].b_w  = (const float*)d_in[p++]; pr[i].b_b  = (const float*)d_in[p++];
        pr[i].c_w  = (const float*)d_in[p++]; pr[i].c_b  = (const float*)d_in[p++];
    }
    const float* nr_w   = (const float*)d_in[p++];
    const float* nr_b   = (const float*)d_in[p++];
    const float* head_w = (const float*)d_in[p++];    // (1536, 768)
    const float* head_b = (const float*)d_in[p++];    // (1536)
    const float* gum    = (const float*)d_in[p++];    // (3, 16, 392, 2)

    // --- workspace partition (floats) ---
    const int MT = 16 * 392;                      // 6272 rows
    const long long SZ_HD  = (long long)MT * 768;
    const long long SZ_QKV = (long long)MT * 2304;
    const long long SZ_ATT = 16LL * 392 * 392;    // per-head score buffer

    float* ws     = (float*)d_ws;
    float* h      = ws;                  // persistent activations
    float* pol    = h + SZ_HD;           // persistent policy (B*N)
    float* lnbuf  = pol + 8192;
    float* qkv    = lnbuf + SZ_HD;
    float* att    = qkv + SZ_QKV;
    float* attn_o = att + SZ_ATT;
    float* mlp    = attn_o + SZ_HD;      // fc1 out; also patch xp; feat; V^T
    // time-multiplexed overlays
    float* pred_h = qkv;                 // predictor (qkv idle then)
    float* za     = qkv + SZ_HD;
    float* zb     = za + (long long)MT * 384;
    float* zc     = zb + (long long)MT * 192;
    float* frame  = zc + (long long)MT * 2;
    float* video  = frame + 128LL * 256;
    float* vT     = mlp;                 // V^T (B,768,392); mlp idle during attn

    // --- policy := 1 ---
    k_fill<<<dim3(25), dim3(256), 0, stream>>>(pol, MT, 1.0f);

    // --- patch embedding ---
    k_patch_gather<<<dim3(6272), dim3(256), 0, stream>>>(X, mlp);
    gemm(stream, mlp, 1536, 1, 0, patch_w, 1536, 1, 0, h, 768, 0,
         patch_b, nullptr, 0, 0, MT, 768, 1536, 1.0f, EP_BIAS, 1);
    k_posadd<<<dim3(6272), dim3(256), 0, stream>>>(h, pos);

    int pc = 0;
    for (int i = 0; i < 12; ++i) {
        if (i == 3 || i == 6 || i == 9) {
            // ---- token-keep predictor ----
            k_layernorm<<<dim3(6272), dim3(256), 0, stream>>>(h, lnbuf, pr[pc].ln_w, pr[pc].ln_b);
            gemm(stream, lnbuf, 768, 1, 0, pr[pc].in_w, 768, 1, 0, pred_h, 768, 0,
                 pr[pc].in_b, nullptr, 0, 0, MT, 768, 768, 1.0f, EP_BIAS | EP_GELU, 1);
            k_frame_pool<<<dim3(128), dim3(256), 0, stream>>>(pred_h, pol, frame);
            k_video_pool<<<dim3(16),  dim3(256), 0, stream>>>(pred_h, pol, video);
            k_feat<<<dim3(6272), dim3(256), 0, stream>>>(pred_h, frame, video, mlp);
            gemm(stream, mlp, 768, 1, 0, pr[pc].a_w, 768, 1, 0, za, 384, 0,
                 pr[pc].a_b, nullptr, 0, 0, MT, 384, 768, 1.0f, EP_BIAS | EP_GELU, 1);
            gemm(stream, za, 384, 1, 0, pr[pc].b_w, 384, 1, 0, zb, 192, 0,
                 pr[pc].b_b, nullptr, 0, 0, MT, 192, 384, 1.0f, EP_BIAS | EP_GELU, 1);
            gemm(stream, zb, 192, 1, 0, pr[pc].c_w, 192, 1, 0, zc, 2, 0,
                 pr[pc].c_b, nullptr, 0, 0, MT, 2, 192, 1.0f, EP_BIAS, 1);
            k_decide<<<dim3(25), dim3(256), 0, stream>>>(zc, gum + (long long)pc * MT * 2, pol, MT);
            ++pc;
        }

        // ---- attention ----
        k_layernorm<<<dim3(6272), dim3(256), 0, stream>>>(h, lnbuf, bl[i].ln1_w, bl[i].ln1_b);
        gemm(stream, lnbuf, 768, 1, 0, bl[i].qkv_w, 768, 1, 0, qkv, 2304, 0,
             bl[i].qkv_b, nullptr, 0, 0, MT, 2304, 768, 1.0f, EP_BIAS, 1);
        k_vtrans<<<dim3(16 * 768), dim3(256), 0, stream>>>(qkv, vT);
        for (int hh = 0; hh < 12; ++hh) {
            // scores: att[b][n][m] = 0.125 * q_n . k_m     (batched over b)
            gemm(stream, qkv + hh * 64,        2304, 1, 392 * 2304,
                         qkv + 768 + hh * 64,  2304, 1, 392 * 2304,
                         att, 392, 392 * 392, nullptr, nullptr, 0, 0,
                         392, 392, 64, 0.125f, 0, 16);
            k_attn_softmax<<<dim3(392, 16), dim3(128), 0, stream>>>(att, pol, 392);
            // out[b][n][hh*64+d] = sum_m a[n,m] * vT[d,m]   (unit-k fast path)
            gemm(stream, att, 392, 1, 392 * 392,
                         vT + hh * 64 * 392, 392, 1, 768 * 392,
                         attn_o + hh * 64, 768, 392 * 768, nullptr, nullptr, 0, 0,
                         392, 64, 392, 1.0f, 0, 16);
        }
        gemm(stream, attn_o, 768, 1, 0, bl[i].proj_w, 768, 1, 0, h, 768, 0,
             bl[i].proj_b, h, 768, 0, MT, 768, 768, 1.0f, EP_BIAS | EP_RES, 1);

        // ---- MLP ----
        k_layernorm<<<dim3(6272), dim3(256), 0, stream>>>(h, lnbuf, bl[i].ln2_w, bl[i].ln2_b);
        gemm(stream, lnbuf, 768, 1, 0, bl[i].fc1_w, 768, 1, 0, mlp, 3072, 0,
             bl[i].fc1_b, nullptr, 0, 0, MT, 3072, 768, 1.0f, EP_BIAS | EP_GELU, 1);
        gemm(stream, mlp, 3072, 1, 0, bl[i].fc2_w, 3072, 1, 0, h, 768, 0,
             bl[i].fc2_b, h, 768, 0, MT, 768, 3072, 1.0f, EP_BIAS | EP_RES, 1);
    }

    // ---- reconstruction head ----
    k_layernorm<<<dim3(6272), dim3(256), 0, stream>>>(h, lnbuf, nr_w, nr_b);
    gemm(stream, lnbuf, 768, 1, 0, head_w, 768, 1, 0, (float*)d_out, 1536, 0,
         head_b, nullptr, 0, 0, MT, 1536, 768, 1.0f, EP_BIAS, 1);
}
